// GraphAttentionNetwork_50878182588496
// MI455X (gfx1250) — matrix operand order repaired and checked by reading, not compiled, not run on hardware
//
#include <hip/hip_runtime.h>

typedef __attribute__((ext_vector_type(16))) _Float16 v16h;
typedef __attribute__((ext_vector_type(8)))  _Float16 v8h;
typedef __attribute__((ext_vector_type(8)))  float    v8f;

#define NODES 20000
#define EDGES 640000
#define ETOT  (EDGES + NODES)
#define HD 64
#define SDIM 12
#define LEAK 0.2f

// ---------------------------------------------------------------- atomic max (float)
__device__ __forceinline__ void atomic_max_f32(float* a, float v) {
    if (v >= 0.0f) atomicMax((int*)a, __float_as_int(v));
    else           atomicMin((unsigned int*)a, __float_as_uint(v));
}

// ---------------------------------------------------------------- weight prep
// wt16[l][o][i] = gatW_l[i][o] (f16, transposed)   3 x 64 x 64
// wc16[o][t*64+i] = conv_w[o][i][t] (f16)          64 x 192
__global__ void k_prep(const float* __restrict__ g0, const float* __restrict__ g1,
                       const float* __restrict__ g2, const float* __restrict__ cw,
                       _Float16* __restrict__ wt16, _Float16* __restrict__ wc16) {
    int t = blockIdx.x * 256 + threadIdx.x;
    if (t < 3 * 4096) {
        int l = t / 4096, r = t % 4096;
        int o = r / 64, i = r % 64;
        const float* w = (l == 0) ? g0 : (l == 1) ? g1 : g2;
        wt16[l * 4096 + o * 64 + i] = (_Float16)w[i * 64 + o];
    } else {
        int u = t - 3 * 4096;
        if (u < 64 * 192) {
            int o = u / 192, k = u % 192;
            int tt = k / 64, i = k % 64;
            wc16[o * 192 + tt * 64 + i] = (_Float16)cw[o * 192 + i * 3 + tt];
        }
    }
}

// ---------------------------------------------------------------- input projection
// h16[n][k] = sum_s x[s][n] * pw[s][k] + pb[k]   (x already offset to batch)
__global__ void k_proj(const float* __restrict__ x, const float* __restrict__ pw,
                       const float* __restrict__ pb, _Float16* __restrict__ h16) {
    int t = blockIdx.x * 256 + threadIdx.x;
    if (t >= NODES * HD) return;
    int n = t >> 6, k = t & 63;
    float acc = pb[k];
#pragma unroll
    for (int s = 0; s < SDIM; ++s) acc += x[s * NODES + n] * pw[s * HD + k];
    h16[t] = (_Float16)acc;
}

// ---------------------------------------------------------------- WMMA GEMM
// C[20000][64] = A_f16 (row m = A + m*64, K=KSTEPS*32 contiguous halves)
//                x BT_f16 ([64 cols][K] row-major, pre-transposed B)
// One wave per (ntile, chunk of MCH m-tiles). B fragments live in registers
// across the whole M chunk; next A row prefetched (WGP scope) during WMMA.
// Strides are compile-time so stores fold into immediate offsets.
template<int KSTEPS, int MCH>
__global__ void k_gemm_t(const _Float16* __restrict__ A,
                         const _Float16* __restrict__ BT,
                         float* __restrict__ C) {
    constexpr int K = KSTEPS * 32;
    constexpr int RS = 64;     // A row stride (halves)
    constexpr int LDC = 64;    // C row stride (floats)
    const int lane = threadIdx.x;
    const int half = lane >> 4, l16 = lane & 15;
    const int ntile = blockIdx.x;
    const int m0 = blockIdx.y * (MCH * 16);

    // B layout used here (contiguous k, not ISA table order): lane col = l16, elems 0..15 = K[kb+16h .. +16)
    const _Float16* bcol = BT + (ntile * 16 + l16) * K;
    v16h bfrag[KSTEPS];
#pragma unroll
    for (int ks = 0; ks < KSTEPS; ++ks)
        bfrag[ks] = *(const v16h*)(bcol + ks * 32 + 16 * half);

    const _Float16* arow = A + (m0 + l16) * RS;
#pragma unroll
    for (int mi = 0; mi < MCH; ++mi) {
        if (mi + 1 < MCH) __builtin_prefetch(arow + 16 * RS, 0, 3);  // next A row, WGP scope
        v8f c = {};
#pragma unroll
        for (int ks = 0; ks < KSTEPS; ++ks) {
            v16h a;
            // A layout matching B: elems 0..15 = K[kb+16h .. +16)
            a = *(const v16h*)(arow + ks * 32 + 16 * half);
            c = __builtin_amdgcn_wmma_f32_16x16x32_f16(false, a, false, bfrag[ks],
                                                       (short)0, c, false, false);
        }
        // ISA C layout: VGPR g -> row 8h+g, col l16  (immediate-offset stores)
        float* crow = C + (m0 + mi * 16 + 8 * half) * LDC + ntile * 16 + l16;
#pragma unroll
        for (int g = 0; g < 8; ++g) crow[g * LDC] = c[g];
        arow += 16 * RS;
    }
}

// ---------------------------------------------------------------- attention coefficients
__global__ void k_attn(const float* __restrict__ xe, const float* __restrict__ asrc,
                       const float* __restrict__ adst, float* __restrict__ es,
                       float* __restrict__ ed, int hshift) {
    int H = 1 << hshift, C = 64 >> hshift;
    int t = blockIdx.x * 256 + threadIdx.x;
    if (t >= NODES * H) return;
    int hh = t & (H - 1);
    int n = t >> hshift;
    const float* xr = xe + (size_t)n * 64 + hh * C;
    float s = 0.f, d = 0.f;
    for (int c = 0; c < C; ++c) {
        s += xr[c] * asrc[hh * C + c];
        d += xr[c] * adst[hh * C + c];
    }
    es[t] = s; ed[t] = d;
}

// ---------------------------------------------------------------- per-layer init
__global__ void k_init(float* __restrict__ m, float* __restrict__ s,
                       float* __restrict__ outacc, int H) {
    int t = blockIdx.x * 256 + threadIdx.x;
    if (t < NODES * H) { m[t] = __int_as_float(0xff800000); s[t] = 0.f; }
    if (t < NODES * 64) outacc[t] = 0.f;
}

__device__ __forceinline__ void edge_endpoints(const int* __restrict__ ei, int e,
                                               int& src, int& dst) {
    if (e < EDGES) { src = ei[e]; dst = ei[EDGES + e]; }
    else           { src = dst = e - EDGES; }   // self loop
}

// ---------------------------------------------------------------- softmax pass 1: max
__global__ void k_edge_max(const int* __restrict__ ei, const float* __restrict__ es,
                           const float* __restrict__ ed, float* __restrict__ m,
                           int hshift) {
    int H = 1 << hshift;
    long t = (long)blockIdx.x * 256 + threadIdx.x;
    if (t >= (long)ETOT * H) return;
    int e = (int)(t >> hshift), hh = (int)t & (H - 1);
    int src, dst; edge_endpoints(ei, e, src, dst);
    float f = es[src * H + hh] + ed[dst * H + hh];
    f = f > 0.f ? f : LEAK * f;
    atomic_max_f32(&m[dst * H + hh], f);
}

// ---------------------------------------------------------------- softmax pass 2: sum
__global__ void k_edge_sum(const int* __restrict__ ei, const float* __restrict__ es,
                           const float* __restrict__ ed, const float* __restrict__ m,
                           float* __restrict__ s, int hshift) {
    int H = 1 << hshift;
    long t = (long)blockIdx.x * 256 + threadIdx.x;
    if (t >= (long)ETOT * H) return;
    int e = (int)(t >> hshift), hh = (int)t & (H - 1);
    int src, dst; edge_endpoints(ei, e, src, dst);
    int idx = dst * H + hh;
    float f = es[src * H + hh] + ed[idx];
    f = f > 0.f ? f : LEAK * f;
    atomicAdd(&s[idx], __expf(f - m[idx]));
}

// ---------------------------------------------------------------- softmax pass 3: aggregate
// One thread per (edge, 4-feature group): b128 gather of xe + 4 scalar atomics.
__global__ void k_edge_agg(const int* __restrict__ ei, const float* __restrict__ es,
                           const float* __restrict__ ed, const float* __restrict__ m,
                           const float* __restrict__ s, const float* __restrict__ xe,
                           float* __restrict__ outacc, int hshift) {
    int H = 1 << hshift, cshift = 6 - hshift;
    long t = (long)blockIdx.x * 256 + threadIdx.x;
    if (t >= (long)ETOT * 16) return;
    int e = (int)(t >> 4), k = ((int)t & 15) * 4;
    int hh = k >> cshift;                       // 4-feature group never straddles a head
    int src, dst; edge_endpoints(ei, e, src, dst);
    int idx = dst * H + hh;
    float f = es[src * H + hh] + ed[idx];
    f = f > 0.f ? f : LEAK * f;
    float alpha = __expf(f - m[idx]) / s[idx];
    float4 xs = *(const float4*)(xe + (size_t)src * 64 + k);
    float* o = outacc + (size_t)dst * 64 + k;
    atomicAdd(o + 0, alpha * xs.x);
    atomicAdd(o + 1, alpha * xs.y);
    atomicAdd(o + 2, alpha * xs.z);
    atomicAdd(o + 3, alpha * xs.w);
}

// ---------------------------------------------------------------- finalize layer (bias, elu, -> f16)
__global__ void k_finalize(const float* __restrict__ outacc, const float* __restrict__ bias,
                           _Float16* __restrict__ dst16, int applyElu) {
    int t = blockIdx.x * 256 + threadIdx.x;
    if (t >= NODES * 64) return;
    int k = t & 63;
    float v = outacc[t] + bias[k];
    if (applyElu) v = v > 0.f ? v : (__expf(v) - 1.f);
    dst16[t] = (_Float16)v;
}

// ---------------------------------------------------------------- zero pad rows for conv
__global__ void k_zero_pad(_Float16* __restrict__ hp16) {
    int t = threadIdx.x;  // 64 threads
    hp16[t] = (_Float16)0.f;
    hp16[(size_t)(NODES + 1) * 64 + t] = (_Float16)0.f;
}

// ---------------------------------------------------------------- conv epilogue + output head
__global__ void k_outproj(const float* __restrict__ hc, const float* __restrict__ cb,
                          const float* __restrict__ ow, const float* __restrict__ ob,
                          float* __restrict__ out) {
    int n = blockIdx.x * 256 + threadIdx.x;
    if (n >= NODES) return;
    float o0 = ob[0], o1 = ob[1], o2 = ob[2];
    for (int i = 0; i < 64; ++i) {
        float v = hc[(size_t)n * 64 + i] + cb[i];
        v = v > 0.f ? v : 0.f;                    // relu after conv
        o0 += v * ow[i * 3 + 0];
        o1 += v * ow[i * 3 + 1];
        o2 += v * ow[i * 3 + 2];
    }
    out[n] = o0; out[NODES + n] = o1; out[2 * NODES + n] = o2;
}

// ================================================================ host
extern "C" void kernel_launch(void* const* d_in, const int* in_sizes, int n_in,
                              void* d_out, int out_size, void* d_ws, size_t ws_size,
                              hipStream_t stream) {
    const float* x   = (const float*)d_in[0];
    const int*   ei  = (const int*)d_in[1];
    const float* pw  = (const float*)d_in[2];
    const float* pb  = (const float*)d_in[3];
    const float* gw[3]  = {(const float*)d_in[4],  (const float*)d_in[8],  (const float*)d_in[12]};
    const float* gas[3] = {(const float*)d_in[5],  (const float*)d_in[9],  (const float*)d_in[13]};
    const float* gad[3] = {(const float*)d_in[6],  (const float*)d_in[10], (const float*)d_in[14]};
    const float* gb[3]  = {(const float*)d_in[7],  (const float*)d_in[11], (const float*)d_in[15]};
    const float* cw = (const float*)d_in[16];
    const float* cb = (const float*)d_in[17];
    const float* ow = (const float*)d_in[18];
    const float* ob = (const float*)d_in[19];
    float* out = (float*)d_out;
    (void)in_sizes; (void)n_in; (void)out_size; (void)ws_size;

    char* ws = (char*)d_ws;
    size_t off = 0;
    auto alloc = [&](size_t bytes) -> void* {
        void* p = ws + off;
        off += (bytes + 255) & ~(size_t)255;
        return p;
    };
    _Float16* wt16 = (_Float16*)alloc((size_t)3 * 4096 * 2);
    _Float16* wc16 = (_Float16*)alloc((size_t)64 * 192 * 2);
    _Float16* h16  = (_Float16*)alloc((size_t)NODES * 64 * 2);
    _Float16* hp16 = (_Float16*)alloc((size_t)(NODES + 2) * 64 * 2);
    float* xe   = (float*)alloc((size_t)NODES * 64 * 4);
    float* outa = (float*)alloc((size_t)NODES * 64 * 4);
    float* hcv  = (float*)alloc((size_t)NODES * 64 * 4);
    float* es   = (float*)alloc((size_t)NODES * 8 * 4);
    float* ed   = (float*)alloc((size_t)NODES * 8 * 4);
    float* mx   = (float*)alloc((size_t)NODES * 8 * 4);
    float* sm   = (float*)alloc((size_t)NODES * 8 * 4);

    const int TB = 256;
    const int gNK = (NODES * 64 + TB - 1) / TB;
    const int MCH = 5;                                // m-tiles per wave
    const dim3 gemmGrid(4, (NODES / 16) / MCH);       // 4 ntiles x 250 chunks

    k_prep<<<(3 * 4096 + 64 * 192 + TB - 1) / TB, TB, 0, stream>>>(
        gw[0], gw[1], gw[2], cw, wt16, wc16);

    for (int b = 0; b < 4; ++b) {
        k_proj<<<gNK, TB, 0, stream>>>(x + (size_t)b * SDIM * NODES, pw, pb, h16);

        for (int l = 0; l < 3; ++l) {
            int hshift = (l < 2) ? 3 : 0;
            int H = 1 << hshift;
            long eH = (long)ETOT * H;
            // node transform: xe = h16 @ W_l   (WMMA, K=64)
            k_gemm_t<2, MCH><<<gemmGrid, 32, 0, stream>>>(h16, wt16 + l * 4096, xe);
            k_attn<<<(NODES * H + TB - 1) / TB, TB, 0, stream>>>(xe, gas[l], gad[l], es, ed, hshift);
            k_init<<<gNK, TB, 0, stream>>>(mx, sm, outa, H);
            k_edge_max<<<(int)((eH + TB - 1) / TB), TB, 0, stream>>>(ei, es, ed, mx, hshift);
            k_edge_sum<<<(int)((eH + TB - 1) / TB), TB, 0, stream>>>(ei, es, ed, mx, sm, hshift);
            k_edge_agg<<<(int)(((long)ETOT * 16 + TB - 1) / TB), TB, 0, stream>>>(
                ei, es, ed, mx, sm, xe, outa, hshift);
            if (l < 2) {
                k_finalize<<<gNK, TB, 0, stream>>>(outa, gb[l], h16, 1);
            } else {
                k_zero_pad<<<1, 64, 0, stream>>>(hp16);
                k_finalize<<<gNK, TB, 0, stream>>>(outa, gb[2], hp16 + 64, 0);
            }
        }
        // conv1d(k=3) as WMMA GEMM over padded rows: K = 3*64 = 192 contiguous halves
        k_gemm_t<6, MCH><<<gemmGrid, 32, 0, stream>>>(hp16, wc16, hcv);
        // relu + bias + output head
        k_outproj<<<(NODES + TB - 1) / TB, TB, 0, stream>>>(hcv, cb, ow, ob,
                                                            out + (size_t)b * 3 * NODES);
    }
}
